// HAB_54975581388990
// MI455X (gfx1250) — matrix operand order repaired and checked by reading, not compiled, not run on hardware
//
#include <hip/hip_runtime.h>

typedef __attribute__((ext_vector_type(16))) _Float16 v16h;
typedef __attribute__((ext_vector_type(8)))  float    v8f;

static constexpr int NT   = 65536;  // tokens (256*256)
static constexpr int C    = 180;
static constexpr int CP   = 192;    // padded channels
static constexpr int NH   = 6;
static constexpr int HD   = 30;
static constexpr int HDP  = 32;
static constexpr int HIDV = 720;
static constexpr int HIDP = 768;
static constexpr int QKVN = 540;
static constexpr int QKVNP= 576;    // 3*192

// ---------------- CDNA5 async global->LDS copies -----------------------------

#if defined(__HIP_DEVICE_COMPILE__)
#if __has_builtin(__builtin_amdgcn_global_load_async_to_lds_b128) && \
    __has_builtin(__builtin_amdgcn_s_wait_asynccnt)
#define HAB_ASYNC_LDS 1
#endif
#endif

#ifdef HAB_ASYNC_LDS
typedef int ai4 __attribute__((vector_size(16)));
typedef __attribute__((address_space(1))) ai4* ai4_gptr;
typedef __attribute__((address_space(3))) ai4* ai4_lptr;
#endif

// copy 16 bytes (8 f16) global -> LDS
__device__ __forceinline__ void async_copy16(const _Float16* g, _Float16* l) {
#ifdef HAB_ASYNC_LDS
  __builtin_amdgcn_global_load_async_to_lds_b128((ai4_gptr)(g), (ai4_lptr)(l), 0, 0);
#else
  *(float4*)l = *(const float4*)g;
#endif
}
__device__ __forceinline__ void async_join3() {   // <=3 outstanding
#ifdef HAB_ASYNC_LDS
  __builtin_amdgcn_s_wait_asynccnt(3);
#endif
}
__device__ __forceinline__ void async_join0() {   // all done
#ifdef HAB_ASYNC_LDS
  __builtin_amdgcn_s_wait_asynccnt(0);
#endif
}

// ---------------- WMMA helpers (contiguous k = 16*(lane/16) + e in A and B, wave32) -------

__device__ __forceinline__ v8f wmma32(v16h a, v16h b, v8f c) {
  return __builtin_amdgcn_wmma_f32_16x16x32_f16(false, a, false, b, (short)0, c,
                                                false, false);
}

// A fragment: 16x32 (MxK), row-major source, leading dim ld (elements)
__device__ __forceinline__ v16h ld_a_frag(const _Float16* s, int ld) {
  int lane = threadIdx.x & 31;
  const _Float16* p = s + (lane & 15) * ld + ((lane & 16) ? 16 : 0);
  v16h a;
#pragma unroll
  for (int e = 0; e < 16; ++e) a[e] = p[e];
  return a;
}

// B fragment: 32x16 (KxN), source row-major B[k][n] with leading dim ld
__device__ __forceinline__ v16h ld_b_frag(const _Float16* s, int ld) {
  int lane = threadIdx.x & 31;
  const _Float16* p = s + (lane & 15) + ((lane & 16) ? 16 * ld : 0);
  v16h b;
#pragma unroll
  for (int e = 0; e < 16; ++e) b[e] = p[e * ld];
  return b;
}

// B fragment from transposed source: src row-major [N][K] (ld), B[k][n]=src[n][k]
__device__ __forceinline__ v16h ld_bT_frag(const _Float16* s, int ld) {
  int lane = threadIdx.x & 31;
  const _Float16* p = s + (lane & 15) * ld + ((lane & 16) ? 16 : 0);
  v16h b;
#pragma unroll
  for (int e = 0; e < 16; ++e) b[e] = p[e];
  return b;
}

__device__ __forceinline__ float gelu_f(float x) {
  return 0.5f * x * (1.0f + erff(x * 0.70710678118654752f));
}

// ---------------- weight packing ---------------------------------------------

__global__ void pack_lin(const float* __restrict__ src, _Float16* __restrict__ dst,
                         int K, int N, int KP, int NP) {
  int idx = blockIdx.x * 256 + threadIdx.x;
  if (idx >= KP * NP) return;
  int k = idx / NP, n = idx % NP;
  dst[idx] = (k < K && n < N) ? (_Float16)src[k * N + n] : (_Float16)0.0f;
}

// OIHW conv weights -> [tap][IP][OP] f16
__global__ void pack_conv(const float* __restrict__ src, _Float16* __restrict__ dst,
                          int O, int I, int IP, int OP) {
  int idx = blockIdx.x * 256 + threadIdx.x;
  if (idx >= 9 * IP * OP) return;
  int tap = idx / (IP * OP);
  int r = idx % (IP * OP);
  int i = r / OP, o = r % OP;
  int ty = tap / 3, tx = tap % 3;
  dst[idx] = (i < I && o < O) ? (_Float16)src[((o * I + i) * 3 + ty) * 3 + tx]
                              : (_Float16)0.0f;
}

// ---------------- LayerNorm ---------------------------------------------------

__global__ __launch_bounds__(256) void ln_kernel(const float* __restrict__ x,
                                                 const float* __restrict__ g,
                                                 const float* __restrict__ b,
                                                 float* __restrict__ outF,
                                                 _Float16* __restrict__ outH) {
  int wave = threadIdx.x >> 5, lane = threadIdx.x & 31;
  int row = blockIdx.x * 8 + wave;
  float v[6];
  float s1 = 0.0f;
#pragma unroll
  for (int i = 0; i < 6; ++i) {
    int idx = lane + 32 * i;
    float t = (idx < C) ? x[(size_t)row * C + idx] : 0.0f;
    v[i] = t; s1 += t;
  }
#pragma unroll
  for (int off = 16; off > 0; off >>= 1) s1 += __shfl_xor(s1, off, 32);
  float mu = s1 * (1.0f / C);
  float s2 = 0.0f;
#pragma unroll
  for (int i = 0; i < 6; ++i) {
    int idx = lane + 32 * i;
    if (idx < C) { float d = v[i] - mu; s2 += d * d; }
  }
#pragma unroll
  for (int off = 16; off > 0; off >>= 1) s2 += __shfl_xor(s2, off, 32);
  float inv = rsqrtf(s2 * (1.0f / C) + 1e-5f);
#pragma unroll
  for (int i = 0; i < 6; ++i) {
    int idx = lane + 32 * i;
    if (idx < C) {
      float n = (v[i] - mu) * inv * g[idx] + b[idx];
      if (outF) outF[(size_t)row * C + idx] = n;
      outH[(size_t)row * CP + idx] = (_Float16)n;
    } else {
      outH[(size_t)row * CP + idx] = (_Float16)0.0f;
    }
  }
}

// ---------------- generic WMMA GEMM ------------------------------------------
// block: 256 thr (8 waves), tile 128(M) x 64(N), async double-buffered LDS
// MODE: 0=qkv 1=fc1-gelu 2=proj 3=fc2

struct GP {
  const _Float16* A; const _Float16* B; const float* bias;
  int lda, ldb, K, n_valid;
  _Float16* outH; int ldo;
  float* outF;
  const float* resX; const float* convY; const float* caV;
};

template <int MODE>
__global__ __launch_bounds__(256) void gemm_kernel(GP p) {
  __shared__ __align__(16) _Float16 As[2][128 * 32];
  __shared__ __align__(16) _Float16 Bs[2][32 * 64];
  const int tid = threadIdx.x, wave = tid >> 5, lane = tid & 31;
  const int wm = wave & 3, wn = wave >> 2;
  const int rowBase = blockIdx.y * 128, colBase = blockIdx.x * 64;
  const int aidx = tid * 16, ar = aidx >> 5, ac = aidx & 31;
  const int bidx = tid * 8, br = bidx >> 6, bc = bidx & 63;
  const _Float16* gA = p.A + (size_t)(rowBase + ar) * p.lda + ac;
  const _Float16* gB = p.B + (size_t)br * p.ldb + colBase + bc;

  auto issue = [&](int ks, int buf) {
    async_copy16(gA + (ks << 5), &As[buf][aidx]);
    async_copy16(gA + (ks << 5) + 8, &As[buf][aidx + 8]);
    async_copy16(gB + (size_t)(ks << 5) * p.ldb, &Bs[buf][bidx]);
  };

  v8f acc[2][2] = {};
  const int ksteps = p.K >> 5;
  issue(0, 0);
  for (int ks = 0; ks < ksteps; ++ks) {
    const int cur = ks & 1;
    if (ks + 1 < ksteps) { issue(ks + 1, cur ^ 1); async_join3(); }
    else                 { async_join0(); }
    __syncthreads();
    v16h a0 = ld_a_frag(&As[cur][(wm * 32) * 32], 32);
    v16h a1 = ld_a_frag(&As[cur][(wm * 32 + 16) * 32], 32);
    v16h b0 = ld_b_frag(&Bs[cur][wn * 32], 64);
    v16h b1 = ld_b_frag(&Bs[cur][wn * 32 + 16], 64);
    acc[0][0] = wmma32(a0, b0, acc[0][0]);
    acc[0][1] = wmma32(a0, b1, acc[0][1]);
    acc[1][0] = wmma32(a1, b0, acc[1][0]);
    acc[1][1] = wmma32(a1, b1, acc[1][1]);
    __syncthreads();
  }
  const int mlane = lane & 15, hi8 = (lane & 16) ? 8 : 0;
#pragma unroll
  for (int i = 0; i < 2; ++i)
#pragma unroll
    for (int j = 0; j < 2; ++j)
#pragma unroll
      for (int v = 0; v < 8; ++v) {
        int r = rowBase + wm * 32 + i * 16 + v + hi8;
        int n = colBase + wn * 32 + j * 16 + mlane;
        float val = acc[i][j][v];
        if constexpr (MODE == 0) {          // qkv: remap to padded per-head layout
          if (n < p.n_valid) {
            val += p.bias[n];
            int three = n / 180, rem = n % 180, hh = rem / 30, d = rem % 30;
            if (three == 0) val *= 0.18257418583505537f;  // 30^-0.5
            p.outH[(size_t)r * QKVNP + three * 192 + hh * 32 + d] = (_Float16)val;
          }
        } else if constexpr (MODE == 1) {   // fc1 + exact GELU -> f16
          _Float16 o = (_Float16)0.0f;
          if (n < p.n_valid) o = (_Float16)gelu_f(val + p.bias[n]);
          p.outH[(size_t)r * p.ldo + n] = o;
        } else if constexpr (MODE == 2) {   // proj + window-reverse/roll + residual
          if (n < C) {
            int win = r >> 8, tok = r & 255;
            int yy = (((win >> 4) << 4) + (tok >> 4) + 8) & 255;
            int xx = (((win & 15) << 4) + (tok & 15) + 8) & 255;
            size_t t = (size_t)(yy * 256 + xx);
            p.outF[t * C + n] = p.resX[t * C + n] + val + p.bias[n] +
                                p.convY[t * C + n] * p.caV[n] * 0.01f;
          }
        } else {                            // fc2 + residual -> final output
          if (n < C)
            p.outF[(size_t)r * C + n] = p.resX[(size_t)r * C + n] + val + p.bias[n];
        }
      }
}

// ---------------- conv as 9-tap shifted GEMM (async double-buffered) ---------

template <bool CONV1>
__global__ __launch_bounds__(256) void conv_gemm_kernel(
    const _Float16* __restrict__ A, int lda, const _Float16* __restrict__ Wp,
    const float* __restrict__ bias, _Float16* __restrict__ outH,
    float* __restrict__ outF, int Kin, int Np, const _Float16* __restrict__ zbuf) {
  __shared__ __align__(16) _Float16 As[2][128 * 32];
  __shared__ __align__(16) _Float16 Bs[2][32 * 64];
  const int tid = threadIdx.x, wave = tid >> 5, lane = tid & 31;
  const int wm = wave & 3, wn = wave >> 2;
  const int rowBase = blockIdx.y * 128, colBase = blockIdx.x * 64;
  const int aidx = tid * 16, ar = aidx >> 5, ac = aidx & 31;
  const int bidx = tid * 8, br = bidx >> 6, bc = bidx & 63;
  const int pix = rowBase + ar, py = pix >> 8, px = pix & 255;
  const int ksteps = Kin >> 5;
  const int nstages = 9 * ksteps;

  auto issue = [&](int s, int buf) {
    int tap = s / ksteps, ks = s - tap * ksteps, k0 = ks << 5;
    int ys = py + tap / 3 - 1, xs = px + tap % 3 - 1;
    bool ok = (unsigned)ys < 256u && (unsigned)xs < 256u;
    const _Float16* ga = ok ? A + (size_t)(ys * 256 + xs) * lda + k0 + ac : zbuf;
    async_copy16(ga, &As[buf][aidx]);
    async_copy16(ok ? ga + 8 : zbuf, &As[buf][aidx + 8]);
    async_copy16(Wp + ((size_t)tap * Kin + k0 + br) * Np + colBase + bc,
                 &Bs[buf][bidx]);
  };

  v8f acc[2][2] = {};
  issue(0, 0);
  for (int s = 0; s < nstages; ++s) {
    const int cur = s & 1;
    if (s + 1 < nstages) { issue(s + 1, cur ^ 1); async_join3(); }
    else                 { async_join0(); }
    __syncthreads();
    v16h a0 = ld_a_frag(&As[cur][(wm * 32) * 32], 32);
    v16h a1 = ld_a_frag(&As[cur][(wm * 32 + 16) * 32], 32);
    v16h b0 = ld_b_frag(&Bs[cur][wn * 32], 64);
    v16h b1 = ld_b_frag(&Bs[cur][wn * 32 + 16], 64);
    acc[0][0] = wmma32(a0, b0, acc[0][0]);
    acc[0][1] = wmma32(a0, b1, acc[0][1]);
    acc[1][0] = wmma32(a1, b0, acc[1][0]);
    acc[1][1] = wmma32(a1, b1, acc[1][1]);
    __syncthreads();
  }
  const int mlane = lane & 15, hi8 = (lane & 16) ? 8 : 0;
#pragma unroll
  for (int i = 0; i < 2; ++i)
#pragma unroll
    for (int j = 0; j < 2; ++j)
#pragma unroll
      for (int v = 0; v < 8; ++v) {
        int r = rowBase + wm * 32 + i * 16 + v + hi8;
        int n = colBase + wn * 32 + j * 16 + mlane;
        float val = acc[i][j][v];
        if constexpr (CONV1) {
          outH[(size_t)r * 64 + n] =
              (n < 60) ? (_Float16)gelu_f(val + bias[n]) : (_Float16)0.0f;
        } else {
          if (n < C) outF[(size_t)r * C + n] = val + bias[n];
        }
      }
}

// ---------------- channel attention ------------------------------------------

__global__ __launch_bounds__(256) void pool_kernel(const float* __restrict__ y2,
                                                   float* __restrict__ pooled) {
  int ch = blockIdx.x;
  float s = 0.0f;
  for (int p = threadIdx.x; p < NT; p += 256) s += y2[(size_t)p * C + ch];
#pragma unroll
  for (int off = 16; off > 0; off >>= 1) s += __shfl_xor(s, off, 32);
  __shared__ float red[8];
  if ((threadIdx.x & 31) == 0) red[threadIdx.x >> 5] = s;
  __syncthreads();
  if (threadIdx.x == 0) {
    float t = 0.0f;
    for (int i = 0; i < 8; ++i) t += red[i];
    pooled[ch] = t * (1.0f / NT);
  }
}

__global__ void ca_kernel(const float* __restrict__ pooled,
                          const float* __restrict__ w1, const float* __restrict__ b1,
                          const float* __restrict__ w2, const float* __restrict__ b2,
                          float* __restrict__ ca) {
  __shared__ float hid[6];
  int t = threadIdx.x;
  if (t < 6) {
    float s = b1[t];
    for (int c = 0; c < C; ++c) s += pooled[c] * w1[t * C + c];
    hid[t] = fmaxf(s, 0.0f);
  }
  __syncthreads();
  if (t < C) {
    float s = b2[t];
    for (int j = 0; j < 6; ++j) s += hid[j] * w2[t * 6 + j];
    ca[t] = 1.0f / (1.0f + __expf(-s));
  }
}

// ---------------- windowed attention (one block per (head, window)) ----------

__global__ __launch_bounds__(512) void attn_kernel(
    const _Float16* __restrict__ qkvp, const int* __restrict__ rpi,
    const float* __restrict__ rpbt, const float* __restrict__ mask,
    _Float16* __restrict__ attn_out) {
  const int head = blockIdx.x, win = blockIdx.y;
  const int tid = threadIdx.x, wave = tid >> 5, lane = tid & 31;
  const int wy = win >> 4, wx = win & 15;
  __shared__ __align__(16) _Float16 Ks[256 * 32];
  __shared__ __align__(16) _Float16 Vs[256 * 32];
  __shared__ __align__(16) _Float16 Sst[16][16 * 32];

  // async-stage K,V (hd padded to 32) for this (window, head) into LDS
#pragma unroll
  for (int e = 0; e < 2; ++e) {
    int idx8 = tid + e * 512;                 // 1024 chunks of 8 f16
    int tok = idx8 >> 2, d8 = (idx8 & 3) * 8;
    int yy = ((wy << 4) + (tok >> 4) + 8) & 255;
    int xx = ((wx << 4) + (tok & 15) + 8) & 255;
    const _Float16* bp =
        qkvp + (size_t)(yy * 256 + xx) * QKVNP + head * HDP + d8;
    async_copy16(bp + 192, Ks + idx8 * 8);
    async_copy16(bp + 384, Vs + idx8 * 8);
  }
  async_join0();
  __syncthreads();

  const int mlane = lane & 15, hi8 = (lane & 16) ? 8 : 0;
  // Q fragment (16 queries of this wave) gathered straight from global
  int qrow = wave * 16 + mlane;
  int qy = ((wy << 4) + (qrow >> 4) + 8) & 255;
  int qx = ((wx << 4) + (qrow & 15) + 8) & 255;
  const _Float16* qp =
      qkvp + (size_t)(qy * 256 + qx) * QKVNP + head * HDP + ((lane & 16) ? 16 : 0);
  v16h aq;
#pragma unroll
  for (int e = 0; e < 16; ++e) aq[e] = qp[e];

  // scores = q k^T  (16 x 256) for this wave's query rows
  v8f sc[16];
#pragma unroll
  for (int nt = 0; nt < 16; ++nt) {
    v8f z = {};
    v16h bk = ld_bT_frag(Ks + nt * 16 * 32, 32);
    sc[nt] = wmma32(aq, bk, z);
  }

  // + relative position bias + shift mask
#pragma unroll
  for (int nt = 0; nt < 16; ++nt)
#pragma unroll
    for (int v = 0; v < 8; ++v) {
      int qr = wave * 16 + v + hi8;
      int kr = nt * 16 + mlane;
      float bb = rpbt[rpi[qr * 256 + kr] * NH + head] +
                 mask[((size_t)win * 256 + qr) * 256 + kr];
      sc[nt][v] += bb;
    }

  // softmax over 256 keys per query row (16-lane group reductions)
#pragma unroll
  for (int v = 0; v < 8; ++v) {
    float mx = -1e30f;
#pragma unroll
    for (int nt = 0; nt < 16; ++nt) mx = fmaxf(mx, sc[nt][v]);
#pragma unroll
    for (int off = 1; off < 16; off <<= 1) mx = fmaxf(mx, __shfl_xor(mx, off, 32));
    float sum = 0.0f;
#pragma unroll
    for (int nt = 0; nt < 16; ++nt) {
      float e = __expf(sc[nt][v] - mx);
      sc[nt][v] = e; sum += e;
    }
#pragma unroll
    for (int off = 1; off < 16; off <<= 1) sum += __shfl_xor(sum, off, 32);
    float inv = 1.0f / sum;
#pragma unroll
    for (int nt = 0; nt < 16; ++nt) sc[nt][v] *= inv;
  }

  // out = softmax(scores) @ V : stage scores per-wave as f16 A-fragments
  v8f o0 = {}, o1 = {};
  _Float16* st = &Sst[wave][0];
#pragma unroll
  for (int kt = 0; kt < 8; ++kt) {
#pragma unroll
    for (int half = 0; half < 2; ++half) {
      int nt = kt * 2 + half;
#pragma unroll
      for (int v = 0; v < 8; ++v)
        st[(v + hi8) * 32 + half * 16 + mlane] = (_Float16)sc[nt][v];
    }
    v16h as = ld_a_frag(st, 32);
    v16h b0 = ld_b_frag(Vs + kt * 32 * 32, 32);
    v16h b1 = ld_b_frag(Vs + kt * 32 * 32 + 16, 32);
    o0 = wmma32(as, b0, o0);
    o1 = wmma32(as, b1, o1);
  }

  // store window-order rows, head-packed cols (pads zeroed by memset)
#pragma unroll
  for (int v = 0; v < 8; ++v) {
    int qr = wave * 16 + v + hi8;
    size_t rb = ((size_t)win * 256 + qr) * CP + head * HD;
    attn_out[rb + mlane] = (_Float16)o0[v];
    int d1 = 16 + mlane;
    if (d1 < HD) attn_out[rb + d1] = (_Float16)o1[v];
  }
}

// ---------------- host launch -------------------------------------------------

extern "C" void kernel_launch(void* const* d_in, const int* in_sizes, int n_in,
                              void* d_out, int out_size, void* d_ws, size_t ws_size,
                              hipStream_t stream) {
  const float* x      = (const float*)d_in[0];
  const int*   rpi    = (const int*)d_in[3];
  const float* amask  = (const float*)d_in[4];
  const float* n1g    = (const float*)d_in[5];
  const float* n1b    = (const float*)d_in[6];
  const float* qkv_w  = (const float*)d_in[7];
  const float* qkv_b  = (const float*)d_in[8];
  const float* rpbt   = (const float*)d_in[9];
  const float* proj_w = (const float*)d_in[10];
  const float* proj_b = (const float*)d_in[11];
  const float* c1w    = (const float*)d_in[12];
  const float* c1b    = (const float*)d_in[13];
  const float* c2w    = (const float*)d_in[14];
  const float* c2b    = (const float*)d_in[15];
  const float* ca1w   = (const float*)d_in[16];
  const float* ca1b   = (const float*)d_in[17];
  const float* ca2w   = (const float*)d_in[18];
  const float* ca2b   = (const float*)d_in[19];
  const float* n2g    = (const float*)d_in[20];
  const float* n2b    = (const float*)d_in[21];
  const float* fc1w   = (const float*)d_in[22];
  const float* fc1b   = (const float*)d_in[23];
  const float* fc2w   = (const float*)d_in[24];
  const float* fc2b   = (const float*)d_in[25];
  float* out = (float*)d_out;

  char* ws = (char*)d_ws;
  size_t off = 0;
  auto A = [&](size_t bytes) -> char* {
    char* p = ws + off;
    off += (bytes + 255) & ~(size_t)255;
    return p;
  };

  const size_t qkvp_bytes = (size_t)NT * QKVNP * 2;   // 75.5 MB
  const size_t attn_bytes = (size_t)NT * CP * 2;      // 25.2 MB
  char* region0      = A(qkvp_bytes + attn_bytes);    // qkvp+attn_out; later m
  _Float16* qkvp     = (_Float16*)region0;
  _Float16* attn_o   = (_Float16*)(region0 + qkvp_bytes);
  _Float16* mbuf     = (_Float16*)region0;            // NT*768*2 == sum, aliased
  float*    xn       = (float*)A((size_t)NT * C * 4);
  _Float16* xnp      = (_Float16*)A((size_t)NT * CP * 2);
  float*    x2       = (float*)A((size_t)NT * C * 4);
  _Float16* xm       = (_Float16*)A((size_t)NT * CP * 2);
  _Float16* y1       = (_Float16*)A((size_t)NT * 64 * 2);
  float*    y2       = (float*)A((size_t)NT * C * 4);
  float*    pooled   = (float*)A(256);
  float*    cav      = (float*)A(256);
  _Float16* zbuf     = (_Float16*)A(256);             // zeroed OOB source chunk
  _Float16* qkv_wp   = (_Float16*)A((size_t)CP * QKVNP * 2);
  _Float16* proj_wp  = (_Float16*)A((size_t)CP * CP * 2);
  _Float16* fc1_wp   = (_Float16*)A((size_t)CP * HIDP * 2);
  _Float16* fc2_wp   = (_Float16*)A((size_t)HIDP * CP * 2);
  _Float16* conv1p   = (_Float16*)A((size_t)9 * CP * 64 * 2);
  _Float16* conv2p   = (_Float16*)A((size_t)9 * 64 * CP * 2);

  // zero padded qkv slots + attn_out pad columns + OOB chunk (every call)
  (void)hipMemsetAsync(region0, 0, qkvp_bytes + attn_bytes, stream);
  (void)hipMemsetAsync(zbuf, 0, 256, stream);

  // pack weights (f32 -> padded f16)
  pack_lin<<<(CP * QKVNP + 255) / 256, 256, 0, stream>>>(qkv_w, qkv_wp, C, QKVN, CP, QKVNP);
  pack_lin<<<(CP * CP + 255) / 256, 256, 0, stream>>>(proj_w, proj_wp, C, C, CP, CP);
  pack_lin<<<(CP * HIDP + 255) / 256, 256, 0, stream>>>(fc1w, fc1_wp, C, HIDV, CP, HIDP);
  pack_lin<<<(HIDP * CP + 255) / 256, 256, 0, stream>>>(fc2w, fc2_wp, HIDV, C, HIDP, CP);
  pack_conv<<<(9 * CP * 64 + 255) / 256, 256, 0, stream>>>(c1w, conv1p, 60, C, CP, 64);
  pack_conv<<<(9 * 64 * CP + 255) / 256, 256, 0, stream>>>(c2w, conv2p, C, 60, 64, CP);

  // LN1
  ln_kernel<<<NT / 8, 256, 0, stream>>>(x, n1g, n1b, xn, xnp);

  // CAB conv branch
  conv_gemm_kernel<true ><<<dim3(1, NT / 128), 256, 0, stream>>>(xnp, CP, conv1p, c1b, y1, nullptr, CP, 64, zbuf);
  conv_gemm_kernel<false><<<dim3(3, NT / 128), 256, 0, stream>>>(y1, 64, conv2p, c2b, nullptr, y2, 64, CP, zbuf);
  pool_kernel<<<C, 256, 0, stream>>>(y2, pooled);
  ca_kernel<<<1, 192, 0, stream>>>(pooled, ca1w, ca1b, ca2w, ca2b, cav);

  // qkv projection
  {
    GP p{}; p.A = xnp; p.B = qkv_wp; p.bias = qkv_b;
    p.lda = CP; p.ldb = QKVNP; p.K = CP; p.n_valid = QKVN;
    p.outH = qkvp; p.ldo = QKVNP;
    gemm_kernel<0><<<dim3(QKVNP / 64, NT / 128), 256, 0, stream>>>(p);
  }

  // shifted-window attention
  attn_kernel<<<dim3(NH, 256), 512, 0, stream>>>(qkvp, rpi, rpbt, amask, attn_o);

  // proj + window reverse/roll + residual (x + attn + 0.01*conv*ca)
  {
    GP p{}; p.A = attn_o; p.B = proj_wp; p.bias = proj_b;
    p.lda = CP; p.ldb = CP; p.K = CP; p.n_valid = C;
    p.outF = x2; p.resX = x; p.convY = y2; p.caV = cav;
    gemm_kernel<2><<<dim3(CP / 64, NT / 128), 256, 0, stream>>>(p);
  }

  // LN2
  ln_kernel<<<NT / 8, 256, 0, stream>>>(x2, n2g, n2b, nullptr, xm);

  // MLP fc1 (GELU) -> mbuf (aliases dead qkv/attn region)
  {
    GP p{}; p.A = xm; p.B = fc1_wp; p.bias = fc1b;
    p.lda = CP; p.ldb = HIDP; p.K = CP; p.n_valid = HIDV;
    p.outH = mbuf; p.ldo = HIDP;
    gemm_kernel<1><<<dim3(HIDP / 64, NT / 128), 256, 0, stream>>>(p);
  }

  // MLP fc2 + residual -> d_out
  {
    GP p{}; p.A = mbuf; p.B = fc2_wp; p.bias = fc2b;
    p.lda = HIDP; p.ldb = CP; p.K = HIDP; p.n_valid = C;
    p.outF = out; p.resX = x2;
    gemm_kernel<3><<<dim3(CP / 64, NT / 128), 256, 0, stream>>>(p);
  }
}
